// MiniMaxText01LightningAttention_52673478918580
// MI455X (gfx1250) — matrix operand order repaired and checked by reading, not compiled, not run on hardware
//
#include <hip/hip_runtime.h>

#ifndef __has_builtin
#define __has_builtin(x) 0
#endif
#if __has_builtin(__builtin_amdgcn_global_load_async_to_lds_b128)
#define HAVE_ASYNC 1
#else
#define HAVE_ASYNC 0
#endif

// ---------------- CDNA5 WMMA fragment helpers (wave32, 16x16x32 bf16) -------
typedef __attribute__((ext_vector_type(16))) __bf16 v16bf;
typedef __attribute__((ext_vector_type(8)))  __bf16 v8bf;
typedef __attribute__((ext_vector_type(8)))  float  v8f;
typedef int v4i __attribute__((vector_size(16)));

#define DEV __device__ __forceinline__

DEV v8f vzero8() {
  v8f r;
#pragma unroll
  for (int i = 0; i < 8; i++) r[i] = 0.f;
  return r;
}

DEV v8f wmma_bf16(v16bf a, v16bf b, v8f c) {
  return __builtin_amdgcn_wmma_f32_16x16x32_bf16(
      /*neg_a=*/false, a, /*neg_b=*/false, b,
      /*c_mod=*/(short)0, c, /*reuse_a=*/false, /*reuse_b=*/false);
}

// A fragment: 16x32 bf16, lane L -> row M = L&15; koff = (L>=16)?8:0;
// elems 0..7 = K kb+koff.., elems 8..15 = K kb+16+koff..   (ISA 7.12.2)
DEV v16bf load_a_frag(const __bf16* p, int ld, int mb, int kb, int lane) {
  int m  = mb + (lane & 15);
  int ko = kb + ((lane >> 4) << 3);
  const __bf16* q = p + (size_t)m * ld + ko;
  v8bf lo = *(const v8bf*)q;
  v8bf hi = *(const v8bf*)(q + 16);
  return __builtin_shufflevector(lo, hi, 0,1,2,3,4,5,6,7,8,9,10,11,12,13,14,15);
}

// B fragment: 32x16 bf16, lane L -> col N = nb+(L&15); koff = (L>=16)?8:0;
// elems 0..7 = K kb+koff.., elems 8..15 = K kb+16+koff..
DEV v16bf load_b_frag(const __bf16* p, int ld, int kb, int nb, int lane) {
  const __bf16* q = p + (size_t)(kb + ((lane >> 4) << 3)) * ld + nb + (lane & 15);
  v16bf r;
#pragma unroll
  for (int i = 0; i < 16; i++) r[i] = q[(size_t)(i + ((i >> 3) << 3)) * ld];
  return r;
}

DEV v16bf scale_frag(v16bf a, float f) {
  v16bf r;
#pragma unroll
  for (int i = 0; i < 16; i++) r[i] = (__bf16)((float)a[i] * f);
  return r;
}

// 16-byte async global->LDS copy (gfx1250 GLOBAL_LOAD_ASYNC_TO_LDS_B128).
// Builtin signature (from hipcc diagnostic): (v4i32 as1* src, v4i32 as3* dst,
// imm offset, imm cpol).
#if HAVE_ASYNC
DEV void async_cp16(const __bf16* gsrc, __bf16* ldst) {
  __builtin_amdgcn_global_load_async_to_lds_b128(
      (__attribute__((address_space(1))) v4i*)gsrc,
      (__attribute__((address_space(3))) v4i*)ldst, 0, 0);
}
DEV void wait_async0() {
#if __has_builtin(__builtin_amdgcn_s_wait_asynccnt)
  __builtin_amdgcn_s_wait_asynccnt(0);
#else
  asm volatile("s_wait_asynccnt 0" ::: "memory");
#endif
}
#endif

// ---------------- problem constants ----------------------------------------
constexpr int BB = 2, NN = 8192, HID = 2048, HH = 16, DD = 128, BLK = 256;
constexpr int MROWS = BB * NN;       // 16384
constexpr int NB = NN / BLK;         // 32
constexpr int BH = BB * HH;          // 32

// ---------------- fp32 -> bf16 cast ----------------------------------------
__global__ __launch_bounds__(256) void cvt_bf16_kernel(const float* in, __bf16* out, size_t n) {
  size_t i = (size_t)blockIdx.x * 256 + threadIdx.x;
  size_t stride = (size_t)gridDim.x * 256;
  for (; i < n; i += stride) out[i] = (__bf16)in[i];
}

// ---------------- tiled WMMA GEMM, 128x256 tile, 8 waves, K-step 32 ---------
// Double-buffered LDS staging via async global->LDS DMA (ASYNCcnt) when
// available; wave tile 32x128 = 16 WMMAs per k-step.
// EPI 0: silu -> scatter to q/k/v [b,h,n,d] bf16   (N = 6144)
// EPI 1: sigmoid -> gate bf16 [M,2048]
// EPI 2: plain f32 store [M,2048]
template <int EPI>
__global__ __launch_bounds__(256) void wmma_gemm(
    const __bf16* __restrict__ A, const __bf16* __restrict__ Bw,
    int M, int N, int K, float* outf,
    __bf16* qout, __bf16* kout, __bf16* vout, __bf16* gout) {
  __shared__ __bf16 As[2][128 * 40];   // 128 rows x K32, padded stride 40
  __shared__ __bf16 Bs[2][32 * 264];   // 32 K-rows x 256 N, padded stride 264
  const int t = threadIdx.x, lane = t & 31, w = t >> 5;
  const int wm = w & 3, wn = w >> 2;   // 4 m-waves x 2 n-waves
  const int m0 = blockIdx.y * 128, n0 = blockIdx.x * 256;

  v8f acc[2][8];
#pragma unroll
  for (int i = 0; i < 2; i++)
#pragma unroll
    for (int j = 0; j < 8; j++) acc[i][j] = vzero8();

  const int arow = t >> 1, akc = (t & 1) * 16;      // A: 128x32, 32B/thread
  const int brow = t >> 3, bnc = (t & 7) * 32;      // B: 32x256, 64B/thread

  auto stage = [&](int buf, int k0) {
    const __bf16* asrc = A + (size_t)(m0 + arow) * K + k0 + akc;
    __bf16* adst = &As[buf][arow * 40 + akc];
    const __bf16* bsrc = Bw + (size_t)(k0 + brow) * N + n0 + bnc;
    __bf16* bdst = &Bs[buf][brow * 264 + bnc];
#if HAVE_ASYNC
    async_cp16(asrc, adst);
    async_cp16(asrc + 8, adst + 8);
#pragma unroll
    for (int c = 0; c < 4; c++) async_cp16(bsrc + 8 * c, bdst + 8 * c);
#else
    v8bf a0 = *(const v8bf*)asrc, a1 = *(const v8bf*)(asrc + 8);
    *(v8bf*)adst = a0;
    *(v8bf*)(adst + 8) = a1;
#pragma unroll
    for (int c = 0; c < 4; c++) *(v8bf*)(bdst + 8 * c) = *(const v8bf*)(bsrc + 8 * c);
#endif
  };

  const int nk = K / 32;
  stage(0, 0);
#if HAVE_ASYNC
  wait_async0();
#endif
  __syncthreads();

  int buf = 0;
  for (int ki = 0; ki < nk; ki++, buf ^= 1) {
    if (ki + 1 < nk) {
      stage(buf ^ 1, (ki + 1) * 32);  // overlap DMA with WMMA below
      __builtin_prefetch(A + (size_t)(m0 + arow) * K + (ki + 2) * 32, 0, 1);
    }
    v16bf bfr[8];
#pragma unroll
    for (int tn = 0; tn < 8; tn++)
      bfr[tn] = load_b_frag(Bs[buf], 264, 0, wn * 128 + tn * 16, lane);
#pragma unroll
    for (int tm = 0; tm < 2; tm++) {
      v16bf af = load_a_frag(As[buf], 40, wm * 32 + tm * 16, 0, lane);
#pragma unroll
      for (int tn = 0; tn < 8; tn++) acc[tm][tn] = wmma_bf16(af, bfr[tn], acc[tm][tn]);
    }
#if HAVE_ASYNC
    wait_async0();
#endif
    __syncthreads();
  }

  // Epilogue. C layout: VGPR r, lanes 0-15 -> M=r,N=lane; 16-31 -> M=r+8.
#pragma unroll
  for (int tm = 0; tm < 2; tm++) {
#pragma unroll
    for (int tn = 0; tn < 8; tn++) {
      int n  = n0 + wn * 128 + tn * 16 + (lane & 15);
      int mb = m0 + wm * 32 + tm * 16 + ((lane >> 4) << 3);
#pragma unroll
      for (int r = 0; r < 8; r++) {
        int m = mb + r;
        float c = acc[tm][tn][r];
        if constexpr (EPI == 0) {
          float val = c / (1.f + __expf(-c));  // silu
          int b = m >> 13, nseq = m & (NN - 1);
          int which = n >> 11, hd = n & (HID - 1);
          int h = hd >> 7, d = hd & (DD - 1);
          __bf16* dst = (which == 0) ? qout : ((which == 1) ? kout : vout);
          dst[(((size_t)(b * HH + h)) * NN + nseq) * DD + d] = (__bf16)val;
        } else if constexpr (EPI == 1) {
          gout[(size_t)m * HID + n] = (__bf16)(1.f / (1.f + __expf(-c)));
        } else {
          outf[(size_t)m * HID + n] = c;
        }
      }
    }
  }
}

// ---------------- pass A: per-block KV_i = (k_i * k_decay)^T @ v_i ----------
__global__ __launch_bounds__(256) void kv_block_kernel(
    const __bf16* __restrict__ Kt, const __bf16* __restrict__ Vt,
    const float* __restrict__ slope, float* __restrict__ kvout) {
  extern __shared__ char smraw[];
  __bf16* kT = (__bf16*)smraw;  // [128 d][264 m] (padded)
  const int blk = blockIdx.x, bh = blockIdx.y;
  const float s = slope[bh & (HH - 1)];
  const int t = threadIdx.x, lane = t & 31, w = t >> 5;
  const __bf16* kblk = Kt + ((size_t)bh * NN + blk * BLK) * DD;
  const __bf16* vblk = Vt + ((size_t)bh * NN + blk * BLK) * DD;

  for (int i = 0; i < 128; i++) {  // 256*128 elements, transpose + k_decay
    int flat = i * 256 + t;
    int m = flat >> 7, d = flat & (DD - 1);
    float kv = (float)kblk[(size_t)m * DD + d];
    kT[d * 264 + m] = (__bf16)(kv * __expf(-s * (float)(BLK - 1 - m)));
  }
  __syncthreads();

  v8f acc[8];
#pragma unroll
  for (int j = 0; j < 8; j++) acc[j] = vzero8();
  const int drow = w * 16;  // each wave: 16 d-rows x 128 e
  for (int ks = 0; ks < 8; ks++) {  // K = 256
    v16bf bfr[8];                   // batch all global B loads -> one wait
#pragma unroll
    for (int j = 0; j < 8; j++) bfr[j] = load_b_frag(vblk, DD, ks * 32, j * 16, lane);
    v16bf af = load_a_frag(kT, 264, drow, ks * 32, lane);
#pragma unroll
    for (int j = 0; j < 8; j++) acc[j] = wmma_bf16(af, bfr[j], acc[j]);
  }
  float* outp = kvout + ((size_t)bh * NB + blk) * (DD * DD);
  const int nbl = lane & 15, rb = (lane >> 4) << 3;
#pragma unroll
  for (int j = 0; j < 8; j++)
#pragma unroll
    for (int r = 0; r < 8; r++)
      outp[(size_t)(drow + rb + r) * DD + j * 16 + nbl] = acc[j][r];
}

// ---------------- pass B: decay-weighted prefix scan over 32 blocks ---------
__global__ __launch_bounds__(256) void kv_scan_kernel(
    const float* __restrict__ kvin, __bf16* __restrict__ kvpre,
    const float* __restrict__ slope) {
  const int bh = blockIdx.x, t = threadIdx.x;
  const float bdec = __expf(-slope[bh & (HH - 1)] * (float)BLK);
  float acc[64];
#pragma unroll
  for (int i = 0; i < 64; i++) acc[i] = 0.f;
  for (int blk = 0; blk < NB; blk++) {
    const float* src = kvin + ((size_t)bh * NB + blk) * (DD * DD);
    __bf16* dst = kvpre + ((size_t)bh * NB + blk) * (DD * DD);
#pragma unroll
    for (int i = 0; i < 64; i++) {
      int idx = i * 256 + t;
      dst[idx] = (__bf16)acc[i];                 // state entering block blk
      acc[i] = acc[i] * bdec + src[idx];
    }
  }
}

// ---------------- pass C: out_i = (q*qdec)@KVpre + causal(q@k^T)*decay @ v --
__global__ __launch_bounds__(256) void attn_out_kernel(
    const __bf16* __restrict__ Qt, const __bf16* __restrict__ Kt,
    const __bf16* __restrict__ Vt, const __bf16* __restrict__ kvpre,
    const float* __restrict__ slope, float* __restrict__ o) {
  extern __shared__ char smraw[];
  __bf16* kT = (__bf16*)smraw;            // [128][264]
  __bf16* vS = kT + 128 * 264;            // [256][136]
  __bf16* Sb = vS + 256 * 136;            // 8 waves x [32][40]
  const int blk = blockIdx.x, bh = blockIdx.y;
  const int b = bh >> 4, h = bh & (HH - 1);
  const float s = slope[h];
  const int t = threadIdx.x, lane = t & 31, w = t >> 5;
  const size_t base = ((size_t)bh * NN + blk * BLK) * DD;
  const __bf16* qblk = Qt + base;
  const __bf16* kblk = Kt + base;
  const __bf16* vblk = Vt + base;
  const __bf16* kvp = kvpre + ((size_t)bh * NB + blk) * (DD * DD);

  // kick off v staging early: straight 16B copies -> async DMA to LDS
  for (int i = 0; i < 16; i++) {
    int flat = i * 256 + t;
    int m = flat >> 4, e = (flat & 15) * 8;
#if HAVE_ASYNC
    async_cp16(vblk + (size_t)m * DD + e, vS + m * 136 + e);
#else
    *(v8bf*)(vS + m * 136 + e) = *(const v8bf*)(vblk + (size_t)m * DD + e);
#endif
  }

  // cache raw q A-fragments for this wave's 32-row strip (K = 128 -> 4 steps)
  v16bf qf[2][4];
#pragma unroll
  for (int tm = 0; tm < 2; tm++)
#pragma unroll
    for (int ks = 0; ks < 4; ks++)
      qf[tm][ks] = load_a_frag(qblk, DD, w * 32 + tm * 16, ks * 32, lane);

  v8f acc[2][8];
#pragma unroll
  for (int i = 0; i < 2; i++)
#pragma unroll
    for (int j = 0; j < 8; j++) acc[i][j] = vzero8();

  // -------- inter: (q * q_decay) @ KV_prefix --------
  const int l0 = w * 32 + (lane & 15);
  const float qd0 = __expf(-s * (float)(l0 + 1));
  const float qd1 = __expf(-s * (float)(l0 + 16 + 1));
  for (int ks = 0; ks < 4; ks++) {
    v16bf bfr[8];                      // batch global B loads -> one wait
#pragma unroll
    for (int j = 0; j < 8; j++) bfr[j] = load_b_frag(kvp, DD, ks * 32, j * 16, lane);
    v16bf a0 = scale_frag(qf[0][ks], qd0);
    v16bf a1 = scale_frag(qf[1][ks], qd1);
#pragma unroll
    for (int j = 0; j < 8; j++) {
      acc[0][j] = wmma_bf16(a0, bfr[j], acc[0][j]);
      acc[1][j] = wmma_bf16(a1, bfr[j], acc[1][j]);
    }
  }

  // -------- stage k^T into LDS (transpose: element-wise) --------
  for (int i = 0; i < 128; i++) {
    int flat = i * 256 + t;
    int m = flat >> 7, d = flat & (DD - 1);
    kT[d * 264 + m] = kblk[(size_t)m * DD + d];
  }
#if HAVE_ASYNC
  wait_async0();   // own v-tile DMA done
#endif
  __syncthreads();

  // -------- intra: causal (q@k^T)*diag_decay @ v, 32-row strip per wave -----
  __bf16* Sw = Sb + w * (32 * 40);
  for (int mc = 0; mc <= w; mc++) {  // only chunks at/below the diagonal
    v8f sa[2][2];
#pragma unroll
    for (int i = 0; i < 2; i++)
#pragma unroll
      for (int j = 0; j < 2; j++) sa[i][j] = vzero8();
    for (int ks = 0; ks < 4; ks++) {
      v16bf b0 = load_b_frag(kT, 264, ks * 32, mc * 32 + 0,  lane);
      v16bf b1 = load_b_frag(kT, 264, ks * 32, mc * 32 + 16, lane);
      sa[0][0] = wmma_bf16(qf[0][ks], b0, sa[0][0]);
      sa[0][1] = wmma_bf16(qf[0][ks], b1, sa[0][1]);
      sa[1][0] = wmma_bf16(qf[1][ks], b0, sa[1][0]);
      sa[1][1] = wmma_bf16(qf[1][ks], b1, sa[1][1]);
    }
    // causal mask + decay, then spill to LDS as bf16 A-operand
#pragma unroll
    for (int tm = 0; tm < 2; tm++)
#pragma unroll
      for (int tn = 0; tn < 2; tn++) {
        int mcol = mc * 32 + tn * 16 + (lane & 15);
        int rbase = tm * 16 + ((lane >> 4) << 3);
#pragma unroll
        for (int r = 0; r < 8; r++) {
          int lrow = w * 32 + rbase + r;
          int dlt = lrow - mcol;
          float vv = sa[tm][tn][r];
          vv = (dlt >= 0) ? vv * __expf(-s * (float)dlt) : 0.f;
          Sw[(rbase + r) * 40 + tn * 16 + (lane & 15)] = (__bf16)vv;
        }
      }
    // DS ops are in-order within a wave; make the hand-off explicit anyway.
    asm volatile("s_wait_dscnt 0" ::: "memory");
    v16bf sA0 = load_a_frag(Sw, 40, 0,  0, lane);
    v16bf sA1 = load_a_frag(Sw, 40, 16, 0, lane);
#pragma unroll
    for (int j = 0; j < 8; j++) {
      v16bf bv = load_b_frag(vS, 136, mc * 32, j * 16, lane);
      acc[0][j] = wmma_bf16(sA0, bv, acc[0][j]);
      acc[1][j] = wmma_bf16(sA1, bv, acc[1][j]);
    }
  }

  // -------- store o as [b][n][h*128+e] f32 (norm layout) --------
#pragma unroll
  for (int tm = 0; tm < 2; tm++)
#pragma unroll
    for (int j = 0; j < 8; j++) {
      int e = j * 16 + (lane & 15);
      int rbase = tm * 16 + ((lane >> 4) << 3);
#pragma unroll
      for (int r = 0; r < 8; r++) {
        int nseq = blk * BLK + w * 32 + rbase + r;
        o[((size_t)b * NN + nseq) * HID + h * DD + e] = acc[tm][j][r];
      }
    }
}

// ---------------- RMSNorm + sigmoid gate -> bf16 X --------------------------
__global__ __launch_bounds__(256) void rms_gate_kernel(
    const float* __restrict__ o, const __bf16* __restrict__ G,
    const float* __restrict__ rmsw, __bf16* __restrict__ X) {
  __shared__ float red[256];
  const int row = blockIdx.x, t = threadIdx.x;
  const float* orow = o + (size_t)row * HID;
  float p = 0.f;
#pragma unroll
  for (int i = 0; i < 8; i++) {
    float x = orow[t + i * 256];
    p += x * x;
  }
  red[t] = p;
  __syncthreads();
  for (int off = 128; off > 0; off >>= 1) {
    if (t < off) red[t] += red[t + off];
    __syncthreads();
  }
  const float inv = rsqrtf(red[0] / (float)HID + 1e-6f);
#pragma unroll
  for (int i = 0; i < 8; i++) {
    int c = t + i * 256;
    float v = orow[c] * inv * rmsw[c] * (float)G[(size_t)row * HID + c];
    X[(size_t)row * HID + c] = (__bf16)v;
  }
}

// ---------------- host: workspace carving + launch chain --------------------
extern "C" void kernel_launch(void* const* d_in, const int* in_sizes, int n_in,
                              void* d_out, int out_size, void* d_ws, size_t ws_size,
                              hipStream_t stream) {
  const float* hid   = (const float*)d_in[0];
  const float* slope = (const float*)d_in[1];
  const float* wqkv  = (const float*)d_in[2];
  const float* wgate = (const float*)d_in[3];
  const float* wout  = (const float*)d_in[4];
  const float* rmsw  = (const float*)d_in[5];
  float* outp = (float*)d_out;

  char* p = (char*)d_ws;
  auto take = [&](size_t bytes) {
    char* r = p;
    p += (bytes + 255) & ~(size_t)255;
    return r;
  };
  const size_t qkvElems = (size_t)BH * NN * DD;  // 33.5M
  __bf16* hb     = (__bf16*)take((size_t)MROWS * HID * 2);
  __bf16* wqkvb  = (__bf16*)take((size_t)HID * 3 * HID * 2);
  __bf16* wgateb = (__bf16*)take((size_t)HID * HID * 2);
  __bf16* woutb  = (__bf16*)take((size_t)HID * HID * 2);
  __bf16* qb     = (__bf16*)take(qkvElems * 2);
  __bf16* kb     = (__bf16*)take(qkvElems * 2);
  __bf16* vb     = (__bf16*)take(qkvElems * 2);
  __bf16* Gb     = (__bf16*)take((size_t)MROWS * HID * 2);
  float*  kvblocks = (float*)take((size_t)BH * NB * DD * DD * 4);
  __bf16* kvpre    = (__bf16*)take((size_t)BH * NB * DD * DD * 2);
  float*  ob       = (float*)take((size_t)MROWS * HID * 4);
  __bf16* Xb       = (__bf16*)take((size_t)MROWS * HID * 2);

  const int kvLds = 128 * 264 * 2;                                    // 66 KB
  const int pcLds = 128 * 264 * 2 + 256 * 136 * 2 + 8 * 32 * 40 * 2;  // 154 KB
  (void)hipFuncSetAttribute(reinterpret_cast<const void*>(&kv_block_kernel),
                            hipFuncAttributeMaxDynamicSharedMemorySize, kvLds);
  (void)hipFuncSetAttribute(reinterpret_cast<const void*>(&attn_out_kernel),
                            hipFuncAttributeMaxDynamicSharedMemorySize, pcLds);

  // 1) casts to bf16
  cvt_bf16_kernel<<<4096, 256, 0, stream>>>(hid,   hb,     (size_t)MROWS * HID);
  cvt_bf16_kernel<<<2048, 256, 0, stream>>>(wqkv,  wqkvb,  (size_t)HID * 3 * HID);
  cvt_bf16_kernel<<<1024, 256, 0, stream>>>(wgate, wgateb, (size_t)HID * HID);
  cvt_bf16_kernel<<<1024, 256, 0, stream>>>(wout,  woutb,  (size_t)HID * HID);

  // 2) qkv projection + silu + scatter to [b,h,n,d]
  wmma_gemm<0><<<dim3(3 * HID / 256, MROWS / 128), 256, 0, stream>>>(
      hb, wqkvb, MROWS, 3 * HID, HID, nullptr, qb, kb, vb, nullptr);

  // 3) gate projection + sigmoid
  wmma_gemm<1><<<dim3(HID / 256, MROWS / 128), 256, 0, stream>>>(
      hb, wgateb, MROWS, HID, HID, nullptr, nullptr, nullptr, nullptr, Gb);

  // 4) lightning attention, three-pass parallel form
  kv_block_kernel<<<dim3(NB, BH), 256, kvLds, stream>>>(kb, vb, slope, kvblocks);
  kv_scan_kernel<<<BH, 256, 0, stream>>>(kvblocks, kvpre, slope);
  attn_out_kernel<<<dim3(NB, BH), 256, pcLds, stream>>>(qb, kb, vb, kvpre, slope, ob);

  // 5) RMSNorm * gate -> bf16
  rms_gate_kernel<<<MROWS, 256, 0, stream>>>(ob, Gb, rmsw, Xb);

  // 6) output projection -> f32 d_out
  wmma_gemm<2><<<dim3(HID / 256, MROWS / 128), 256, 0, stream>>>(
      Xb, woutb, MROWS, HID, HID, outp, nullptr, nullptr, nullptr, nullptr);
}
